// TransformerEncoderLayer_35416300323410
// MI455X (gfx1250) — matrix operand order repaired and checked by reading, not compiled, not run on hardware
//
#include <hip/hip_runtime.h>

typedef __attribute__((ext_vector_type(16))) __bf16 v16bf;
typedef __attribute__((ext_vector_type(8)))  __bf16 v8bf;
typedef __attribute__((ext_vector_type(8)))  float  v8f;
typedef int v4i __attribute__((vector_size(16)));
typedef __attribute__((address_space(1))) v4i gv4i;
typedef __attribute__((address_space(3))) v4i lv4i;

#define S_DIM 1024
#define B_DIM 8
#define E_DIM 1024
#define H_DIM 16
#define D_DIM 64
#define DFF_DIM 4096

#if __has_builtin(__builtin_amdgcn_global_load_async_to_lds_b128) && \
    __has_builtin(__builtin_amdgcn_s_wait_asynccnt)
#define USE_ASYNC_LDS 1
#else
#define USE_ASYNC_LDS 0
#endif

__device__ inline v16bf make_v16(v8bf lo, v8bf hi) {
  return __builtin_shufflevector(lo, hi, 0,1,2,3,4,5,6,7,8,9,10,11,12,13,14,15);
}

__device__ inline v8f wmma_bf16(v16bf a, v16bf b, v8f c) {
  return __builtin_amdgcn_wmma_f32_16x16x32_bf16(false, a, false, b, (short)0, c, false, false);
}

// 16-byte global -> LDS copy; async (ASYNCcnt) when the toolchain exposes it.
__device__ inline void cp16_g2l(const __bf16* g, __bf16* l) {
#if USE_ASYNC_LDS
  __builtin_amdgcn_global_load_async_to_lds_b128((gv4i*)g, (lv4i*)l, 0, 0);
#else
  *(v8bf*)l = *(const v8bf*)g;
#endif
}
__device__ inline void cp16_wait() {
#if USE_ASYNC_LDS
  __builtin_amdgcn_s_wait_asynccnt(0);
#endif
}

// ---------------------------------------------------------------------------
// fp32 -> bf16 convert
// ---------------------------------------------------------------------------
__global__ __launch_bounds__(256) void cvt_kernel(const float* __restrict__ in,
                                                  __bf16* __restrict__ out, int n) {
  int i = (blockIdx.x * 256 + threadIdx.x) * 4;
  if (i < n) {
    float4 v = *(const float4*)&in[i];
    out[i + 0] = (__bf16)v.x;
    out[i + 1] = (__bf16)v.y;
    out[i + 2] = (__bf16)v.z;
    out[i + 3] = (__bf16)v.w;
  }
}

// ---------------------------------------------------------------------------
// GEMM: C[M,N] = A[M,K] @ W[N,K]^T + bias, optional ReLU, fp32/bf16 outputs.
// Block tile 128x128, BK=64, 128 threads (4 waves, 2x2), wave tile 64x64
// (4x4 accumulators -> 32 WMMA per K-tile per wave, ~1:1 wmma:ds_load_b128).
// A/W are bf16 row-major. LDS stride 72 elems (144B = 9*16B, conflict-free).
// ---------------------------------------------------------------------------
template <int RELU, int WF32, int WBF16>
__global__ __launch_bounds__(128)
void gemm_bf16_kernel(const __bf16* __restrict__ A, const __bf16* __restrict__ W,
                      const float* __restrict__ bias,
                      float* __restrict__ Cf, __bf16* __restrict__ Cb,
                      int M, int N, int K) {
  const int tid  = threadIdx.x;
  const int wave = tid >> 5, lane = tid & 31;
  const int hl = lane >> 4, ln = lane & 15;
  const int wm = wave >> 1, wn = wave & 1;     // 2x2 wave grid
  const int bm = blockIdx.y * 128, bn = blockIdx.x * 128;

  __shared__ __bf16 sA[128 * 72];
  __shared__ __bf16 sB[128 * 72];

  const v8f vzero = {0.f, 0.f, 0.f, 0.f, 0.f, 0.f, 0.f, 0.f};
  v8f acc[4][4];
#pragma unroll
  for (int mi = 0; mi < 4; ++mi)
#pragma unroll
    for (int ni = 0; ni < 4; ++ni) acc[mi][ni] = vzero;

  for (int k0 = 0; k0 < K; k0 += 64) {
    // Stage 128x64 bf16 tiles of A and W; 8 chunks of 16B per thread each.
#pragma unroll
    for (int p = 0; p < 8; ++p) {
      int idx = p * 128 + tid;
      int r = idx >> 3, c = (idx & 7) * 8;
      cp16_g2l(&A[(size_t)(bm + r) * K + k0 + c], &sA[r * 72 + c]);
      cp16_g2l(&W[(size_t)(bn + r) * K + k0 + c], &sB[r * 72 + c]);
    }
    cp16_wait();
    __syncthreads();

#pragma unroll
    for (int kk = 0; kk < 2; ++kk) {
      const int kb = kk * 32;
      v16bf af[4], bf[4];
#pragma unroll
      for (int mi = 0; mi < 4; ++mi) {
        int r = wm * 64 + mi * 16 + ln;
        v8bf lo = *(const v8bf*)&sA[r * 72 + kb + hl * 16];
        v8bf hi = *(const v8bf*)&sA[r * 72 + kb + hl * 16 + 8];
        af[mi] = make_v16(lo, hi);
      }
#pragma unroll
      for (int ni = 0; ni < 4; ++ni) {
        int r = wn * 64 + ni * 16 + ln;
        bf[ni] = *(const v16bf*)&sB[r * 72 + kb + hl * 16];
      }
#pragma unroll
      for (int mi = 0; mi < 4; ++mi)
#pragma unroll
        for (int ni = 0; ni < 4; ++ni)
          acc[mi][ni] = wmma_bf16(af[mi], bf[ni], acc[mi][ni]);
    }
    __syncthreads();
  }

  // Epilogue: C layout elem e -> row = e + 8*hl, col = ln (within 16x16 tile)
#pragma unroll
  for (int mi = 0; mi < 4; ++mi) {
#pragma unroll
    for (int ni = 0; ni < 4; ++ni) {
      const int colg = bn + wn * 64 + ni * 16 + ln;
      const float bia = bias[colg];
#pragma unroll
      for (int e = 0; e < 8; ++e) {
        const int rowg = bm + wm * 64 + mi * 16 + e + 8 * hl;
        float v = acc[mi][ni][e] + bia;
        if (RELU) v = fmaxf(v, 0.f);
        if (WF32) Cf[(size_t)rowg * N + colg] = v;
        if (WBF16) Cb[(size_t)rowg * N + colg] = (__bf16)v;
      }
    }
  }
}

// ---------------------------------------------------------------------------
// Flash attention. qkv: (S,B,3E) bf16. ctx out: (S,B,E) bf16.
// grid = (S/128, B*H). 8 waves/block; wave owns 16 q-rows; K-blocks of 64.
// ---------------------------------------------------------------------------
__global__ __launch_bounds__(256)
void attn_kernel(const __bf16* __restrict__ qkv, __bf16* __restrict__ ctx) {
  const int tid = threadIdx.x, wave = tid >> 5, lane = tid & 31;
  const int hl = lane >> 4, ln = lane & 15;
  const int b = blockIdx.y >> 4, h = blockIdx.y & 15;
  const int qb = blockIdx.x * 128;
  const float sm = 0.125f;  // 1/sqrt(64)

  __shared__ __bf16 sK[64 * 72];      // [key][d]
  __shared__ __bf16 sV[64 * 72];      // transposed: [d][key]
  __shared__ __bf16 sP[8][16 * 72];   // per-wave P tile [qrow][key]

  // Preload Q fragments (A-layout) for this wave's 16 rows, D=64 -> 2 chunks.
  v16bf qf[2];
  {
    const __bf16* qr = qkv + ((size_t)(qb + wave * 16 + ln) * B_DIM + b) * 3072 + h * 64;
#pragma unroll
    for (int kk = 0; kk < 2; ++kk) {
      v8bf lo = *(const v8bf*)&qr[kk * 32 + hl * 16];
      v8bf hi = *(const v8bf*)&qr[kk * 32 + hl * 16 + 8];
      qf[kk] = make_v16(lo, hi);
    }
  }

  float mrow[8], lrow[8];
  const v8f vzero = {0.f, 0.f, 0.f, 0.f, 0.f, 0.f, 0.f, 0.f};
  v8f accv[4];
#pragma unroll
  for (int e = 0; e < 8; ++e) { mrow[e] = -3.0e38f; lrow[e] = 0.f; }
#pragma unroll
  for (int ni = 0; ni < 4; ++ni) accv[ni] = vzero;

  for (int j = 0; j < S_DIM; j += 64) {
    {  // Stage K tile (row-major, async path) and V tile (transposed).
      int r = tid >> 2, c0 = (tid & 3) * 16;
      const __bf16* kr = qkv + ((size_t)(j + r) * B_DIM + b) * 3072 + 1024 + h * 64 + c0;
      cp16_g2l(kr,     &sK[r * 72 + c0]);
      cp16_g2l(kr + 8, &sK[r * 72 + c0 + 8]);

      int key = tid & 63, d0 = (tid >> 6) * 16;
      const __bf16* vr = qkv + ((size_t)(j + key) * B_DIM + b) * 3072 + 2048 + h * 64 + d0;
      v8bf va = *(const v8bf*)vr, vb = *(const v8bf*)(vr + 8);
#pragma unroll
      for (int i = 0; i < 8; ++i) {
        sV[(d0 + i) * 72 + key]     = va[i];
        sV[(d0 + 8 + i) * 72 + key] = vb[i];
      }
      cp16_wait();
    }
    __syncthreads();

    // scores s = Q @ K^T for 16 q-rows x 64 keys
    v8f s[4];
#pragma unroll
    for (int ni = 0; ni < 4; ++ni) s[ni] = vzero;
#pragma unroll
    for (int kk = 0; kk < 2; ++kk)
#pragma unroll
      for (int ni = 0; ni < 4; ++ni) {
        v16bf bf = *(const v16bf*)&sK[(ni * 16 + ln) * 72 + kk * 32 + hl * 16];
        s[ni] = wmma_bf16(qf[kk], bf, s[ni]);
      }

    // Online softmax update (row stats per elem e; reduce within 16-lane half).
    float mnew[8], scale[8], rs[8];
#pragma unroll
    for (int e = 0; e < 8; ++e) {
      float v = fmaxf(fmaxf(s[0][e], s[1][e]), fmaxf(s[2][e], s[3][e])) * sm;
      v = fmaxf(v, __shfl_xor(v, 1));
      v = fmaxf(v, __shfl_xor(v, 2));
      v = fmaxf(v, __shfl_xor(v, 4));
      v = fmaxf(v, __shfl_xor(v, 8));
      mnew[e]  = fmaxf(mrow[e], v);
      scale[e] = __expf(mrow[e] - mnew[e]);
      mrow[e]  = mnew[e];
      rs[e]    = 0.f;
    }
#pragma unroll
    for (int ni = 0; ni < 4; ++ni)
#pragma unroll
      for (int e = 0; e < 8; ++e) {
        float p = __expf(s[ni][e] * sm - mnew[e]);
        rs[e] += p;
        sP[wave][(e + 8 * hl) * 72 + ni * 16 + ln] = (__bf16)p;
      }
#pragma unroll
    for (int e = 0; e < 8; ++e) {
      float v = rs[e];
      v += __shfl_xor(v, 1);
      v += __shfl_xor(v, 2);
      v += __shfl_xor(v, 4);
      v += __shfl_xor(v, 8);
      lrow[e] = lrow[e] * scale[e] + v;
#pragma unroll
      for (int ni = 0; ni < 4; ++ni) accv[ni][e] *= scale[e];
    }

    // ctx += P @ V  (A-frags from per-wave LDS P, B-frags from transposed V)
#pragma unroll
    for (int kk = 0; kk < 2; ++kk) {
      v8bf lo = *(const v8bf*)&sP[wave][ln * 72 + kk * 32 + hl * 16];
      v8bf hi = *(const v8bf*)&sP[wave][ln * 72 + kk * 32 + hl * 16 + 8];
      v16bf pf = make_v16(lo, hi);
#pragma unroll
      for (int ni = 0; ni < 4; ++ni) {
        v16bf vf = *(const v16bf*)&sV[(ni * 16 + ln) * 72 + kk * 32 + hl * 16];
        accv[ni] = wmma_bf16(pf, vf, accv[ni]);
      }
    }
    __syncthreads();
  }

#pragma unroll
  for (int e = 0; e < 8; ++e) {
    const float rinv = 1.f / lrow[e];
    const int row = qb + wave * 16 + e + 8 * hl;
#pragma unroll
    for (int ni = 0; ni < 4; ++ni) {
      const int d = ni * 16 + ln;
      ctx[((size_t)row * B_DIM + b) * E_DIM + h * 64 + d] = (__bf16)(accv[ni][e] * rinv);
    }
  }
}

// ---------------------------------------------------------------------------
// y = LayerNorm(a + b) * g + beta ; one block per row of E=1024.
// ---------------------------------------------------------------------------
__global__ __launch_bounds__(256)
void add_ln_kernel(const float* __restrict__ a, const float* __restrict__ bsrc,
                   const float* __restrict__ g, const float* __restrict__ beta,
                   float* __restrict__ outf, __bf16* __restrict__ outb) {
  const int row = blockIdx.x, tid = threadIdx.x;
  const int wave = tid >> 5, lane = tid & 31, c = tid * 4;
  const float4 av = *(const float4*)&a[(size_t)row * E_DIM + c];
  const float4 bv = *(const float4*)&bsrc[(size_t)row * E_DIM + c];
  const float x0 = av.x + bv.x, x1 = av.y + bv.y, x2 = av.z + bv.z, x3 = av.w + bv.w;
  float s1 = x0 + x1 + x2 + x3;
  float s2 = x0 * x0 + x1 * x1 + x2 * x2 + x3 * x3;
#pragma unroll
  for (int o = 1; o < 32; o <<= 1) { s1 += __shfl_xor(s1, o); s2 += __shfl_xor(s2, o); }

  __shared__ float red1[8], red2[8];
  if (lane == 0) { red1[wave] = s1; red2[wave] = s2; }
  __syncthreads();
  float t1 = 0.f, t2 = 0.f;
#pragma unroll
  for (int i = 0; i < 8; ++i) { t1 += red1[i]; t2 += red2[i]; }

  const float mean = t1 * (1.f / E_DIM);
  const float var  = t2 * (1.f / E_DIM) - mean * mean;
  const float rstd = rsqrtf(var + 1e-5f);
  const float4 gv = *(const float4*)&g[c];
  const float4 be = *(const float4*)&beta[c];
  const float y0 = (x0 - mean) * rstd * gv.x + be.x;
  const float y1 = (x1 - mean) * rstd * gv.y + be.y;
  const float y2 = (x2 - mean) * rstd * gv.z + be.z;
  const float y3 = (x3 - mean) * rstd * gv.w + be.w;
  *(float4*)&outf[(size_t)row * E_DIM + c] = make_float4(y0, y1, y2, y3);
  if (outb) {
    __bf16* o = outb + (size_t)row * E_DIM + c;
    o[0] = (__bf16)y0; o[1] = (__bf16)y1; o[2] = (__bf16)y2; o[3] = (__bf16)y3;
  }
}

// ---------------------------------------------------------------------------
extern "C" void kernel_launch(void* const* d_in, const int* in_sizes, int n_in,
                              void* d_out, int out_size, void* d_ws, size_t ws_size,
                              hipStream_t stream) {
  (void)in_sizes; (void)n_in; (void)out_size;
  const float* src  = (const float*)d_in[0];
  const float* wqkv = (const float*)d_in[1];
  const float* bqkv = (const float*)d_in[2];
  const float* wout = (const float*)d_in[3];
  const float* bout = (const float*)d_in[4];
  const float* ln1g = (const float*)d_in[5];
  const float* ln1b = (const float*)d_in[6];
  const float* ln2g = (const float*)d_in[7];
  const float* ln2b = (const float*)d_in[8];
  const float* w1   = (const float*)d_in[9];
  const float* b1   = (const float*)d_in[10];
  const float* w2   = (const float*)d_in[11];
  const float* b2   = (const float*)d_in[12];

  if (ws_size < 209715200u) return;  // workspace plan needs 200 MB

  char* ws = (char*)d_ws;
  __bf16* wqkv_b = (__bf16*)(ws + 0);          //  6,291,456
  __bf16* wout_b = (__bf16*)(ws + 6291456);    //  2,097,152
  __bf16* w1_b   = (__bf16*)(ws + 8388608);    //  8,388,608
  __bf16* w2_b   = (__bf16*)(ws + 16777216);   //  8,388,608
  __bf16* src_b  = (__bf16*)(ws + 25165824);   // 16,777,216 (reused as ctx)
  __bf16* ctx_b  = src_b;
  __bf16* qkv_b  = (__bf16*)(ws + 41943040);   // 50,331,648
  float*  x_f    = (float*) (ws + 92274688);   // 33,554,432
  __bf16* x_b    = (__bf16*)(ws + 125829120);  // 16,777,216
  __bf16* h1_b   = (__bf16*)(ws + 142606336);  // 67,108,864
  float*  outp   = (float*)d_out;              // reused as attn_out / ff2_out

  // Down-convert weights + src to bf16.
  cvt_kernel<<<3072, 256, 0, stream>>>(wqkv, wqkv_b, 3 * E_DIM * E_DIM);
  cvt_kernel<<<1024, 256, 0, stream>>>(wout, wout_b, E_DIM * E_DIM);
  cvt_kernel<<<4096, 256, 0, stream>>>(w1, w1_b, DFF_DIM * E_DIM);
  cvt_kernel<<<4096, 256, 0, stream>>>(w2, w2_b, E_DIM * DFF_DIM);
  cvt_kernel<<<8192, 256, 0, stream>>>(src, src_b, S_DIM * B_DIM * E_DIM);

  // qkv = src @ Wqkv^T + b   -> bf16
  gemm_bf16_kernel<0, 0, 1><<<dim3(24, 64), 128, 0, stream>>>(
      src_b, wqkv_b, bqkv, (float*)nullptr, qkv_b, 8192, 3072, 1024);
  // attention -> ctx bf16
  attn_kernel<<<dim3(8, 128), 256, 0, stream>>>(qkv_b, ctx_b);
  // attn_out = ctx @ Wout^T + b -> fp32 in d_out
  gemm_bf16_kernel<0, 1, 0><<<dim3(8, 64), 128, 0, stream>>>(
      ctx_b, wout_b, bout, outp, (__bf16*)nullptr, 8192, 1024, 1024);
  // x = LN(src + attn_out) -> fp32 + bf16
  add_ln_kernel<<<8192, 256, 0, stream>>>(src, outp, ln1g, ln1b, x_f, x_b);
  // h1 = relu(x @ W1^T + b1) -> bf16
  gemm_bf16_kernel<1, 0, 1><<<dim3(32, 64), 128, 0, stream>>>(
      x_b, w1_b, b1, (float*)nullptr, h1_b, 8192, 4096, 1024);
  // ff = h1 @ W2^T + b2 -> fp32 in d_out
  gemm_bf16_kernel<0, 1, 0><<<dim3(8, 64), 128, 0, stream>>>(
      h1_b, w2_b, b2, outp, (__bf16*)nullptr, 8192, 1024, 4096);
  // out = LN(x + ff) (in-place on d_out)
  add_ln_kernel<<<8192, 256, 0, stream>>>(x_f, outp, ln2g, ln2b, outp, (__bf16*)nullptr);
}
